// MultiHeadAttn_22926535426578
// MI455X (gfx1250) — matrix operand order repaired and checked by reading, not compiled, not run on hardware
//
#include <hip/hip_runtime.h>

// ---------------------------------------------------------------------------
// MI455X (gfx1250) fused MHA block: bf16 WMMA GEMMs + flash attention + LN.
// wave32, V_WMMA_F32_16X16X32_BF16 for every matmul; double-buffered LDS
// staging; fragments pre-loaded so WMMAs issue back-to-back; load-phase /
// store-phase split staging so global loads clause and waits amortize.
// ---------------------------------------------------------------------------

typedef unsigned short u16;
typedef __attribute__((ext_vector_type(16))) __bf16 v16bf;
typedef __attribute__((ext_vector_type(2)))  __bf16 v2bf;
typedef __attribute__((ext_vector_type(8)))  float  v8f;

union BFrag {            // 16 bf16 = 8 VGPRs, matches WMMA A/B operand
    v16bf v;
    uint4 q[2];
};

static __device__ __forceinline__ u16 f2bf(float x) {   // native v_cvt (RNE)
    union { __bf16 h; u16 u; } c;
    c.h = (__bf16)x;
    return c.u;
}

#if defined(__has_builtin)
#if __has_builtin(__builtin_amdgcn_cvt_pk_bf16_f32)
#define HAVE_CVT_PK_BF16 1
#endif
#endif

static __device__ __forceinline__ unsigned int pack2bf(float lo, float hi) {
#ifdef HAVE_CVT_PK_BF16
    union { v2bf h; unsigned int u; } c;
    c.h = __builtin_amdgcn_cvt_pk_bf16_f32(lo, hi);
    return c.u;
#else
    union { __bf16 h[2]; unsigned int u; } c;
    c.h[0] = (__bf16)lo;
    c.h[1] = (__bf16)hi;
    return c.u;
#endif
}

static __device__ __forceinline__ v8f wmma_bf16(const BFrag& a, const BFrag& b, v8f c) {
    // (neg_a, A, neg_b, B, c_mod, C, reuse_a, reuse_b)
    return __builtin_amdgcn_wmma_f32_16x16x32_bf16(false, a.v, false, b.v,
                                                   (short)0, c, false, false);
}

// ---------------------------------------------------------------------------
// Kernel 1: fp32 -> bf16 weight conversion (one-shot, 2 MB total)
// ---------------------------------------------------------------------------
__global__ __launch_bounds__(256) void cvt_bf16_kernel(const float* __restrict__ src,
                                                       u16* __restrict__ dst, int n) {
    int i = (blockIdx.x * 256 + threadIdx.x) * 4;
    if (i + 3 < n) {
        float4 f = *(const float4*)(src + i);
        uint2 p; p.x = pack2bf(f.x, f.y); p.y = pack2bf(f.z, f.w);
        *(uint2*)(dst + i) = p;
    }
}

// ---------------------------------------------------------------------------
// Kernel 2: C[M,N] = epilogue( A[M,K](f32->bf16) * W[N,K]^T (bf16) )
// Block tile 128x128, 8 waves in 4x2 grid, each wave 32 rows x 64 cols
// (eight 16x16 fp32 accumulators). K step 32, double-buffered LDS.
// Epilogue: optional bias, optional (resid + relu(acc+bias)), dual f32/bf16 out.
// ---------------------------------------------------------------------------
__global__ __launch_bounds__(256) void gemm_bf16_kernel(
    const float* __restrict__ A, const u16* __restrict__ Wb,
    float* __restrict__ Cf, u16* __restrict__ Cb,
    const float* __restrict__ bias, const float* __restrict__ resid,
    int M, int N, int K)
{
    __shared__ u16 As[2][128][32];   // 16 KB, row-major [m][k] bf16
    __shared__ u16 Bs[2][128][32];   // 16 KB, row-major [n][k] bf16

    const int tid  = threadIdx.x;
    const int w    = tid >> 5;
    const int lane = tid & 31;
    const int half = lane >> 4;     // K-half selector (WMMA 16-bit A/B layout)
    const int l16  = lane & 15;
    const int wr   = w & 3;         // 32-row slice
    const int wc   = w >> 2;        // 64-col slice
    const int m0 = blockIdx.y * 128;
    const int n0 = blockIdx.x * 128;

    // stage one K-chunk into LDS buffer bb: issue ALL global loads first
    // (clause + amortized loadcnt), then convert/store to LDS.
    auto stage = [&](int kk, int bb) {
        float4 fa[4];
#pragma unroll
        for (int i = 0; i < 4; ++i) {
            int idx = tid + i * 256;
            int row = idx >> 3, kq = idx & 7;
            fa[i] = *(const float4*)(A + (size_t)(m0 + row) * K + kk + kq * 4);
        }
        uint4 wb2[2];
#pragma unroll
        for (int i = 0; i < 2; ++i) {
            int idx = tid + i * 256;
            int row = idx >> 2, seg = idx & 3;
            wb2[i] = *(const uint4*)(Wb + (size_t)(n0 + row) * K + kk + seg * 8);
        }
#pragma unroll
        for (int i = 0; i < 4; ++i) {
            int idx = tid + i * 256;
            int row = idx >> 3, kq = idx & 7;
            uint2 p; p.x = pack2bf(fa[i].x, fa[i].y); p.y = pack2bf(fa[i].z, fa[i].w);
            *(uint2*)&As[bb][row][kq * 4] = p;
        }
#pragma unroll
        for (int i = 0; i < 2; ++i) {
            int idx = tid + i * 256;
            int row = idx >> 2, seg = idx & 3;
            *(uint4*)&Bs[bb][row][seg * 8] = wb2[i];
        }
    };

    v8f c[2][4] = {};   // eight 16x16 fp32 accumulators (8 VGPRs each)

    stage(0, 0);
    int buf = 0;
    for (int k0 = 0; k0 < K; k0 += 32, buf ^= 1) {
        __syncthreads();
        if (k0 + 32 < K) stage(k0 + 32, buf ^ 1);

        // pre-load all fragments, then issue the 8 WMMAs back-to-back
        BFrag a0, a1, bb[4];
        a0.q[0] = *(const uint4*)&As[buf][wr * 32 + l16][8 * half];
        a0.q[1] = *(const uint4*)&As[buf][wr * 32 + l16][16 + 8 * half];
        a1.q[0] = *(const uint4*)&As[buf][wr * 32 + 16 + l16][8 * half];
        a1.q[1] = *(const uint4*)&As[buf][wr * 32 + 16 + l16][16 + 8 * half];
#pragma unroll
        for (int nt = 0; nt < 4; ++nt) {
            bb[nt].q[0] = *(const uint4*)&Bs[buf][wc * 64 + nt * 16 + l16][8 * half];
            bb[nt].q[1] = *(const uint4*)&Bs[buf][wc * 64 + nt * 16 + l16][16 + 8 * half];
        }
#pragma unroll
        for (int nt = 0; nt < 4; ++nt) {
            c[0][nt] = wmma_bf16(a0, bb[nt], c[0][nt]);
            c[1][nt] = wmma_bf16(a1, bb[nt], c[1][nt]);
        }
    }

    // epilogue: C layout lane=n, VGPR r -> m = r + 8*half
#pragma unroll
    for (int mt = 0; mt < 2; ++mt) {
#pragma unroll
        for (int nt = 0; nt < 4; ++nt) {
            int n = n0 + wc * 64 + nt * 16 + l16;
            float bv = bias ? bias[n] : 0.0f;
#pragma unroll
            for (int r = 0; r < 8; ++r) {
                int m = m0 + wr * 32 + mt * 16 + r + 8 * half;
                float val = c[mt][nt][r] + bv;
                if (resid) val = resid[(size_t)m * N + n] + fmaxf(val, 0.0f);
                if (Cf) Cf[(size_t)m * N + n] = val;
                if (Cb) Cb[(size_t)m * N + n] = f2bf(val);
            }
        }
    }
}

// ---------------------------------------------------------------------------
// Kernel 3: flash attention. Block = (128 q rows, one (b,h)). 8 waves x 16 rows.
// KV chunked by 32 with double-buffered LDS staging; online softmax with
// shfl_xor row reductions in 16-lane half-groups; P transposed C->A layout
// through a per-wave LDS slab.
// ---------------------------------------------------------------------------
__global__ __launch_bounds__(256) void attn_kernel(
    const u16* __restrict__ qh, const u16* __restrict__ kh,
    const u16* __restrict__ vh, const unsigned char* __restrict__ mask,
    float* __restrict__ attn)
{
    constexpr int LQ = 1024, LK = 1024, D = 512, DH = 64;
    __shared__ u16 Ks[2][32][64];     // 8 KB: [kv][feat] bf16
    __shared__ u16 Vt[2][64][32];     // 8 KB: [feat][kv] bf16 (transposed)
    __shared__ u16 Pb[8][16][32];     // 8 KB: per-wave P slab, [m][kv]

    const int tid  = threadIdx.x;
    const int w    = tid >> 5;
    const int lane = tid & 31;
    const int half = lane >> 4;
    const int l16  = lane & 15;
    const int b = blockIdx.y >> 3, h = blockIdx.y & 7;
    const int q0 = blockIdx.x * 128;
    const float scale = 0.044194173824159216f;   // 1/sqrt(512) (dim_out!)

    // Q tile 16x64 -> two A fragments, loaded once
    const int qm = q0 + w * 16 + l16;
    const u16* qrow = qh + ((size_t)(b * LQ + qm)) * D + h * DH;
    BFrag qf0, qf1;
    qf0.q[0] = *(const uint4*)(qrow + 8 * half);
    qf0.q[1] = *(const uint4*)(qrow + 16 + 8 * half);
    qf1.q[0] = *(const uint4*)(qrow + 32 + 8 * half);
    qf1.q[1] = *(const uint4*)(qrow + 48 + 8 * half);

    // cooperative stage of one 32-row KV chunk: both loads first, then stores
    auto stage = [&](int kk, int bb) {
        int r = tid >> 3, sg = tid & 7;
        size_t base = ((size_t)(b * LK + kk + r)) * D + h * DH + sg * 8;
        uint4 kk4 = *(const uint4*)(kh + base);
        uint4 vv  = *(const uint4*)(vh + base);
        *(uint4*)&Ks[bb][r][sg * 8] = kk4;
        const u16* vp = (const u16*)&vv;
#pragma unroll
        for (int j = 0; j < 8; ++j) Vt[bb][sg * 8 + j][r] = vp[j];
    };

    v8f o[4] = {};
    float mi[8], li[8];
#pragma unroll
    for (int r = 0; r < 8; ++r) { mi[r] = -__builtin_inff(); li[r] = 0.0f; }
    const unsigned char* mrow = mask + (size_t)b * LK;

    stage(0, 0);
    int buf = 0;
    for (int kc = 0; kc < LK; kc += 32, buf ^= 1) {
        __syncthreads();
        if (kc + 32 < LK) stage(kc + 32, buf ^ 1);

        // ---- S = Q * K^T : pre-load 4 K fragments, then 4 chained WMMAs
        BFrag kb[4];
        kb[0].q[0] = *(const uint4*)&Ks[buf][l16][8 * half];
        kb[0].q[1] = *(const uint4*)&Ks[buf][l16][16 + 8 * half];
        kb[1].q[0] = *(const uint4*)&Ks[buf][l16][32 + 8 * half];
        kb[1].q[1] = *(const uint4*)&Ks[buf][l16][48 + 8 * half];
        kb[2].q[0] = *(const uint4*)&Ks[buf][16 + l16][8 * half];
        kb[2].q[1] = *(const uint4*)&Ks[buf][16 + l16][16 + 8 * half];
        kb[3].q[0] = *(const uint4*)&Ks[buf][16 + l16][32 + 8 * half];
        kb[3].q[1] = *(const uint4*)&Ks[buf][16 + l16][48 + 8 * half];
        v8f s0 = {}, s1 = {};
        s0 = wmma_bf16(qf0, kb[0], s0);
        s0 = wmma_bf16(qf1, kb[1], s0);
        s1 = wmma_bf16(qf0, kb[2], s1);
        s1 = wmma_bf16(qf1, kb[3], s1);

        // ---- scale + mask (mask==true -> -inf column)
        const bool msk0 = mrow[kc + l16] != 0;
        const bool msk1 = mrow[kc + 16 + l16] != 0;
#pragma unroll
        for (int r = 0; r < 8; ++r) {
            s0[r] = msk0 ? -__builtin_inff() : s0[r] * scale;
            s1[r] = msk1 ? -__builtin_inff() : s1[r] * scale;
        }

        // ---- online softmax per row (rows live per 16-lane half-group)
#pragma unroll
        for (int r = 0; r < 8; ++r) {
            float rm = fmaxf(s0[r], s1[r]);
            rm = fmaxf(rm, __shfl_xor(rm, 1, 32));
            rm = fmaxf(rm, __shfl_xor(rm, 2, 32));
            rm = fmaxf(rm, __shfl_xor(rm, 4, 32));
            rm = fmaxf(rm, __shfl_xor(rm, 8, 32));
            float mnew = fmaxf(mi[r], rm);
            float corr = (mi[r] > -1e37f) ? __expf(mi[r] - mnew) : 0.0f;
            float p0   = (mnew  > -1e37f) ? __expf(s0[r] - mnew) : 0.0f;
            float p1   = (mnew  > -1e37f) ? __expf(s1[r] - mnew) : 0.0f;
            float rs = p0 + p1;
            rs += __shfl_xor(rs, 1, 32);
            rs += __shfl_xor(rs, 2, 32);
            rs += __shfl_xor(rs, 4, 32);
            rs += __shfl_xor(rs, 8, 32);
            li[r] = li[r] * corr + rs;
            mi[r] = mnew;
            o[0][r] *= corr; o[1][r] *= corr; o[2][r] *= corr; o[3][r] *= corr;
            // spill P (C layout) to per-wave slab as bf16
            int prow = r + 8 * half;
            Pb[w][prow][l16]      = f2bf(p0);
            Pb[w][prow][16 + l16] = f2bf(p1);
        }

        // ---- O += P * V : reload P in A layout, pre-load 4 V fragments
        BFrag pa;
        pa.q[0] = *(const uint4*)&Pb[w][l16][8 * half];
        pa.q[1] = *(const uint4*)&Pb[w][l16][16 + 8 * half];
        BFrag vb[4];
#pragma unroll
        for (int nt = 0; nt < 4; ++nt) {
            vb[nt].q[0] = *(const uint4*)&Vt[buf][nt * 16 + l16][8 * half];
            vb[nt].q[1] = *(const uint4*)&Vt[buf][nt * 16 + l16][16 + 8 * half];
        }
#pragma unroll
        for (int nt = 0; nt < 4; ++nt) o[nt] = wmma_bf16(pa, vb[nt], o[nt]);
    }

    // ---- finalize: divide by l (0 if fully masked, matching NaN->0)
#pragma unroll
    for (int r = 0; r < 8; ++r) {
        float iv = (li[r] > 0.0f) ? 1.0f / li[r] : 0.0f;
        int m = q0 + w * 16 + r + 8 * half;
        size_t off = ((size_t)(b * LQ + m)) * D + h * DH + l16;
        attn[off +  0] = o[0][r] * iv;
        attn[off + 16] = o[1][r] * iv;
        attn[off + 32] = o[2][r] * iv;
        attn[off + 48] = o[3][r] * iv;
    }
}

// ---------------------------------------------------------------------------
// Kernel 4: out = LayerNorm(a + b?) * g + beta, D = 512, one row per block
// ---------------------------------------------------------------------------
__global__ __launch_bounds__(256) void add_ln_kernel(
    const float* __restrict__ a, const float* __restrict__ b,
    const float* __restrict__ g, const float* __restrict__ be,
    float* __restrict__ out)
{
    __shared__ float red[256];
    const int row = blockIdx.x, tid = threadIdx.x;
    const float* ar = a + (size_t)row * 512;
    float x0 = ar[tid]       + (b ? b[(size_t)row * 512 + tid]       : 0.0f);
    float x1 = ar[tid + 256] + (b ? b[(size_t)row * 512 + tid + 256] : 0.0f);

    red[tid] = x0 + x1;
    __syncthreads();
    for (int o2 = 128; o2 > 0; o2 >>= 1) {
        if (tid < o2) red[tid] += red[tid + o2];
        __syncthreads();
    }
    float mean = red[0] * (1.0f / 512.0f);
    __syncthreads();

    float d0 = x0 - mean, d1 = x1 - mean;
    red[tid] = d0 * d0 + d1 * d1;
    __syncthreads();
    for (int o2 = 128; o2 > 0; o2 >>= 1) {
        if (tid < o2) red[tid] += red[tid + o2];
        __syncthreads();
    }
    float rinv = rsqrtf(red[0] * (1.0f / 512.0f) + 1e-5f);

    out[(size_t)row * 512 + tid]       = d0 * rinv * g[tid]       + be[tid];
    out[(size_t)row * 512 + tid + 256] = d1 * rinv * g[tid + 256] + be[tid + 256];
}

// ---------------------------------------------------------------------------
// Host launcher
// ---------------------------------------------------------------------------
extern "C" void kernel_launch(void* const* d_in, const int* in_sizes, int n_in,
                              void* d_out, int out_size, void* d_ws, size_t ws_size,
                              hipStream_t stream) {
    (void)in_sizes; (void)n_in; (void)out_size; (void)ws_size;
    constexpr int Bb = 8, LQ = 1024, D = 512;
    constexpr int M = Bb * LQ;           // 8192

    const float* q  = (const float*)d_in[0];
    const float* k  = (const float*)d_in[1];
    const float* v  = (const float*)d_in[2];
    const unsigned char* mask = (const unsigned char*)d_in[3];
    const float* Wq = (const float*)d_in[4];
    const float* Wk = (const float*)d_in[5];
    const float* Wv = (const float*)d_in[6];
    const float* Wo = (const float*)d_in[7];
    const float* bo = (const float*)d_in[8];
    const float* g1 = (const float*)d_in[9];
    const float* b1 = (const float*)d_in[10];
    const float* g2 = (const float*)d_in[11];
    const float* b2 = (const float*)d_in[12];
    float* out = (float*)d_out;

    char* ws = (char*)d_ws;
    size_t off = 0;
    auto alloc = [&](size_t bytes) -> void* {
        void* p = ws + off;
        off = (off + bytes + 255) & ~(size_t)255;
        return p;
    };
    u16* wq_bf = (u16*)alloc((size_t)D * D * 2);
    u16* wk_bf = (u16*)alloc((size_t)D * D * 2);
    u16* wv_bf = (u16*)alloc((size_t)D * D * 2);
    u16* wo_bf = (u16*)alloc((size_t)D * D * 2);
    float* qp   = (float*)alloc((size_t)M * D * 4);
    u16*   qhb  = (u16*)  alloc((size_t)M * D * 2);
    u16*   khb  = (u16*)  alloc((size_t)M * D * 2);
    u16*   vhb  = (u16*)  alloc((size_t)M * D * 2);
    float* abuf = (float*)alloc((size_t)M * D * 4);   // attn, later reused as tmp2
    float* out1 = (float*)alloc((size_t)M * D * 4);

    // 1) weights -> bf16
    cvt_bf16_kernel<<<D * D / 1024, 256, 0, stream>>>(Wq, wq_bf, D * D);
    cvt_bf16_kernel<<<D * D / 1024, 256, 0, stream>>>(Wk, wk_bf, D * D);
    cvt_bf16_kernel<<<D * D / 1024, 256, 0, stream>>>(Wv, wv_bf, D * D);
    cvt_bf16_kernel<<<D * D / 1024, 256, 0, stream>>>(Wo, wo_bf, D * D);

    // 2) projections (qp needs fp32 for residual; all need bf16 for attention)
    dim3 gg(D / 128, M / 128);
    gemm_bf16_kernel<<<gg, 256, 0, stream>>>(q, wq_bf, qp,      qhb, nullptr, nullptr, M, D, D);
    gemm_bf16_kernel<<<gg, 256, 0, stream>>>(k, wk_bf, nullptr, khb, nullptr, nullptr, M, D, D);
    gemm_bf16_kernel<<<gg, 256, 0, stream>>>(v, wv_bf, nullptr, vhb, nullptr, nullptr, M, D, D);

    // 3) flash attention -> abuf
    dim3 ga(LQ / 128, Bb * 8);
    attn_kernel<<<ga, 256, 0, stream>>>(qhb, khb, vhb, mask, abuf);

    // 4) out1 = LN(qp + attn)
    add_ln_kernel<<<M, 256, 0, stream>>>(qp, abuf, g1, b1, out1);

    // 5) tmp2 = out1 + relu(out1 @ Wo^T + bo)   (abuf reused as tmp2)
    gemm_bf16_kernel<<<gg, 256, 0, stream>>>(out1, wo_bf, abuf, nullptr, bo, out1, M, D, D);

    // 6) out = LN(tmp2)
    add_ln_kernel<<<M, 256, 0, stream>>>(abuf, nullptr, g2, b2, out);
}
